// BidirectionalLSTM_29987461660966
// MI455X (gfx1250) — matrix operand order repaired and checked by reading, not compiled, not run on hardware
//
#include <hip/hip_runtime.h>
#include <hip/hip_bf16.h>
#include <math.h>

// Problem constants (match reference)
#define BB 32
#define TT 512
#define II 256
#define HH 512
#define GG 2048   // 4*H
#define NWG_REC 32

typedef __attribute__((ext_vector_type(16))) __bf16 v16bf;
typedef __attribute__((ext_vector_type(8)))  float  v8f;
typedef __attribute__((ext_vector_type(4)))  unsigned int v4u;

static_assert(sizeof(v16bf) == 32, "v16bf must be 32 bytes");

// ---------------------------------------------------------------------------
// WMMA fragment loaders (CDNA5 16-bit layouts, ISA 7.12.2)
// A (16x32, row-major M x K): lane = M row (lane&15); lanes<16 hold K 0-7 &
// 16-23, lanes>=16 hold K 8-15 & 24-31 (two contiguous 8-elem chunks).
__device__ __forceinline__ v16bf load_fragA(const __bf16* base, int stride, int lane) {
  const int row  = lane & 15;
  const int koff = (lane & 16) ? 8 : 0;
  const __bf16* p = base + (size_t)row * stride + koff;
  union { v16bf v; v4u u[2]; } t;
  t.u[0] = *(const v4u*)(p);        // K chunk [koff .. koff+7]
  t.u[1] = *(const v4u*)(p + 16);   // K chunk [koff+16 .. koff+23]
  return t.v;
}

// B (32x16, K x N): lane = N column; lanes<16 hold K 0-7 & 16-23, lanes>=16 hold
// K 8-15 & 24-31 (same K order as A). Since W is stored [n][k] row-major, a B-tile
// column n is just elements of W row n -> no transpose needed.
__device__ __forceinline__ v16bf load_fragB(const __bf16* base, int stride, int lane) {
  const int col  = lane & 15;
  const int koff = (lane & 16) ? 8 : 0;
  const __bf16* p = base + (size_t)col * stride + koff;
  union { v16bf v; v4u u[2]; } t;
  t.u[0] = *(const v4u*)(p);
  t.u[1] = *(const v4u*)(p + 16);
  return t.v;
}

__device__ __forceinline__ v8f wmma_bf16(v16bf a, v16bf b, v8f c) {
  // (neg_a, A, neg_b, B, c_mod, C, reuse_a, reuse_b)
  return __builtin_amdgcn_wmma_f32_16x16x32_bf16(false, a, false, b, (short)0, c, false, false);
}

__device__ __forceinline__ float sigm(float x) { return 1.0f / (1.0f + __expf(-x)); }

// ---------------------------------------------------------------------------
// fp32 -> bf16 conversion
__global__ void cvt_bf16_kernel(const float* __restrict__ src, __bf16* __restrict__ dst, int n) {
  int i = blockIdx.x * blockDim.x + threadIdx.x;
  if (i < n) dst[i] = (__bf16)src[i];
}

// zero h/c state + grid-barrier words
__global__ void init_state_kernel(__bf16* h_f, __bf16* h_b, float* c_f, float* c_b,
                                  unsigned* bar) {
  int i = blockIdx.x * blockDim.x + threadIdx.x;
  if (i < BB * HH) {
    h_f[i] = (__bf16)0.0f;
    h_b[i] = (__bf16)0.0f;
    c_f[i] = 0.0f;
    c_b[i] = 0.0f;
  }
  if (i < 2) bar[i] = 0u;
}

// ---------------------------------------------------------------------------
// Input projection for both directions:
//   xproj[dir][t][b][g] = sum_i x[b, (dir? T-1-t : t), i] * W_ih[dir][g,i] + bias[g]
// M = B*T rows of x (row index m = b*T + t), N = 4H, K = I.
// Block: 256 threads = 8 waves; waves tile 4(M) x 2(N); block covers 256(M) x 32(N).
__global__ __launch_bounds__(256)
void xproj_kernel(const __bf16* __restrict__ xb,
                  const __bf16* __restrict__ wih_f, const __bf16* __restrict__ wih_b,
                  const float* __restrict__ bias_f, const float* __restrict__ bias_b,
                  float* __restrict__ xpf, float* __restrict__ xpb) {
  const int dir = blockIdx.z;
  const __bf16* wih = dir ? wih_b : wih_f;
  const float* bias = dir ? bias_b : bias_f;
  float* xproj = dir ? xpb : xpf;

  const int lane  = threadIdx.x & 31;
  const int wid   = threadIdx.x >> 5;   // 0..7
  const int waveM = wid & 3;            // 4 waves along M
  const int waveN = wid >> 2;           // 2 waves along N
  const int mbase = blockIdx.y * 256 + waveM * 64;   // 4 M-tiles per wave
  const int n0    = blockIdx.x * 32 + waveN * 16;    // 1 N-tile per wave

  const float bi = bias[n0 + (lane & 15)];
  v8f acc[4];
  for (int mt = 0; mt < 4; ++mt)
    for (int j = 0; j < 8; ++j) acc[mt][j] = bi;   // bias folded into accumulator

  for (int k = 0; k < II; k += 32) {
    if (k + 32 < II) __builtin_prefetch(wih + (size_t)n0 * II + k + 32, 0, 3);
    v16bf bf = load_fragB(wih + (size_t)n0 * II + k, II, lane);
#pragma unroll
    for (int mt = 0; mt < 4; ++mt) {
      v16bf af = load_fragA(xb + (size_t)(mbase + mt * 16) * II + k, II, lane);
      acc[mt] = wmma_bf16(af, bf, acc[mt]);
    }
  }

  const int ncol  = n0 + (lane & 15);
  const int rbase = (lane & 16) ? 8 : 0;
#pragma unroll
  for (int mt = 0; mt < 4; ++mt) {
    for (int j = 0; j < 8; ++j) {
      int m = mbase + mt * 16 + rbase + j;
      int b = m >> 9;          // / T
      int t = m & (TT - 1);    // % T
      int ts = dir ? (TT - 1 - t) : t;
      xproj[((size_t)ts * BB + b) * GG + ncol] = acc[mt][j];
    }
  }
}

// ---------------------------------------------------------------------------
// Software grid barrier (arrival counter + generation), agent scope.
__device__ __forceinline__ void grid_barrier(unsigned* cnt, unsigned* gen, unsigned ngrp) {
  __syncthreads();
  if (threadIdx.x == 0) {
    unsigned g = __hip_atomic_load(gen, __ATOMIC_RELAXED, __HIP_MEMORY_SCOPE_AGENT);
    unsigned prev = __hip_atomic_fetch_add(cnt, 1u, __ATOMIC_ACQ_REL, __HIP_MEMORY_SCOPE_AGENT);
    if (prev == ngrp - 1u) {
      __hip_atomic_store(cnt, 0u, __ATOMIC_RELAXED, __HIP_MEMORY_SCOPE_AGENT);
      __hip_atomic_fetch_add(gen, 1u, __ATOMIC_RELEASE, __HIP_MEMORY_SCOPE_AGENT);
    } else {
      while (__hip_atomic_load(gen, __ATOMIC_ACQUIRE, __HIP_MEMORY_SCOPE_AGENT) == g) {
        __builtin_amdgcn_s_sleep(8);
      }
    }
  }
  __syncthreads();
}

// ---------------------------------------------------------------------------
// Persistent recurrent kernel: 32 WGs x 256 threads = 256 waves.
// Wave w: dir = w>>7, gate-column tile n0 = (w&127)*16; computes both batch
// tiles (b 0-15, 16-31) -> gates_pre[b][n0..n0+15] = h @ W_hh^T (+xproj).
// Each WG stages its direction's h [32 x 512] bf16 (32 KB) into LDS once per
// step; all 8 waves read A-fragments from LDS (ds path), only the per-wave
// W_hh stream (L2-resident) stays on the vmem path.
// Two grid barriers per timestep (gates ready; h/c updated).
__global__ __launch_bounds__(256)
void lstm_recurrent_kernel(const float* __restrict__ xpf, const float* __restrict__ xpb,
                           const __bf16* __restrict__ whh_f, const __bf16* __restrict__ whh_b,
                           __bf16* h_f, __bf16* h_b, float* c_f, float* c_b,
                           float* gates_f, float* gates_b, float* __restrict__ out,
                           unsigned* bar_cnt, unsigned* bar_gen) {
  __shared__ __bf16 hlds[BB * HH];               // 32 KB (of 320 KB WGP LDS)

  const int tid  = threadIdx.x;
  const int lane = tid & 31;
  const int wid  = blockIdx.x * 8 + (tid >> 5);  // 0..255
  const int dir  = wid >> 7;                     // 0 fwd, 1 bwd (uniform per WG)
  const int n0   = (wid & 127) * 16;

  const __bf16* whh   = dir ? whh_b : whh_f;
  const float*  xproj = dir ? xpb   : xpf;
  __bf16*       h     = dir ? h_b   : h_f;
  float*        gates = dir ? gates_b : gates_f;

  const int ncol   = n0 + (lane & 15);
  const int rowoff = (lane & 16) ? 8 : 0;
  const int gthread = blockIdx.x * 256 + tid;    // 0..8191

  for (int t = 0; t < TT; ++t) {
    // --- stage h (global -> LDS), 2048 x uint4 = 32 KB --------------------
    {
      const v4u* hg = (const v4u*)h;
      v4u* hl = (v4u*)hlds;
#pragma unroll
      for (int i = 0; i < (BB * HH * 2) / 16 / 256; ++i)   // 8 iterations
        hl[tid + i * 256] = hg[tid + i * 256];
    }
    __syncthreads();

    // --- gate GEMM: [32 x 16] patch of h @ W_hh^T -------------------------
    v8f acc0 = {};
    v8f acc1 = {};
#pragma unroll 4
    for (int k = 0; k < HH; k += 32) {
      v16bf bf = load_fragB(whh + (size_t)n0 * HH + k, HH, lane);
      v16bf a0 = load_fragA(hlds + k, HH, lane);                    // b = 0..15
      v16bf a1 = load_fragA(hlds + (size_t)16 * HH + k, HH, lane);  // b = 16..31
      acc0 = wmma_bf16(a0, bf, acc0);
      acc1 = wmma_bf16(a1, bf, acc1);
    }

    // add precomputed x-projection (+bias) and publish pre-activation gates
    const float* xp = xproj + (size_t)t * BB * GG;
#pragma unroll
    for (int j = 0; j < 8; ++j) {
      int b0 = rowoff + j;
      gates[(size_t)b0 * GG + ncol] = acc0[j] + xp[(size_t)b0 * GG + ncol];
      int b1 = 16 + rowoff + j;
      gates[(size_t)b1 * GG + ncol] = acc1[j] + xp[(size_t)b1 * GG + ncol];
    }
    __threadfence();
    grid_barrier(bar_cnt, bar_gen, NWG_REC);

    // --- elementwise LSTM cell update: 2 dirs * B * H = 32768 items -------
    for (int item = gthread; item < 2 * BB * HH; item += NWG_REC * 256) {
      int d   = item >> 14;          // / (B*H)
      int rem = item & 16383;
      int b   = rem >> 9;            // / H
      int j   = rem & (HH - 1);
      float*        cc = d ? c_b : c_f;
      __bf16*       hh = d ? h_b : h_f;
      const float*  gg = d ? gates_b : gates_f;
      float gi = sigm(gg[(size_t)b * GG + j]);
      float gf = sigm(gg[(size_t)b * GG + HH + j]);
      float gc = tanhf(gg[(size_t)b * GG + 2 * HH + j]);
      float go = sigm(gg[(size_t)b * GG + 3 * HH + j]);
      float cn = gf * cc[b * HH + j] + gi * gc;
      float hn = go * tanhf(cn);
      cc[b * HH + j] = cn;
      hh[b * HH + j] = (__bf16)hn;
      int tout = d ? (TT - 1 - t) : t;
      out[((size_t)b * TT + tout) * (2 * HH) + d * HH + j] = hn;
    }
    __threadfence();
    grid_barrier(bar_cnt, bar_gen, NWG_REC);
  }
}

// ---------------------------------------------------------------------------
extern "C" void kernel_launch(void* const* d_in, const int* in_sizes, int n_in,
                              void* d_out, int out_size, void* d_ws, size_t ws_size,
                              hipStream_t stream) {
  (void)in_sizes; (void)n_in; (void)out_size; (void)ws_size;
  const float* x     = (const float*)d_in[0];
  const float* Wih_f = (const float*)d_in[1];
  const float* Whh_f = (const float*)d_in[2];
  const float* b_f   = (const float*)d_in[3];
  const float* Wih_b = (const float*)d_in[4];
  const float* Whh_b = (const float*)d_in[5];
  const float* b_b   = (const float*)d_in[6];
  float* out = (float*)d_out;

  char* base = (char*)d_ws;
  size_t off = 0;
  auto take = [&](size_t bytes) -> char* {
    char* p = base + off;
    off = (off + bytes + 255) & ~(size_t)255;
    return p;
  };

  unsigned* bar  = (unsigned*)take(256);
  __bf16* h_f    = (__bf16*)take((size_t)BB * HH * 2);
  __bf16* h_b    = (__bf16*)take((size_t)BB * HH * 2);
  float*  c_f    = (float*) take((size_t)BB * HH * 4);
  float*  c_b    = (float*) take((size_t)BB * HH * 4);
  float*  gts_f  = (float*) take((size_t)BB * GG * 4);
  float*  gts_b  = (float*) take((size_t)BB * GG * 4);
  __bf16* xb     = (__bf16*)take((size_t)BB * TT * II * 2);
  __bf16* wihf   = (__bf16*)take((size_t)GG * II * 2);
  __bf16* wihb   = (__bf16*)take((size_t)GG * II * 2);
  __bf16* whhf   = (__bf16*)take((size_t)GG * HH * 2);
  __bf16* whhb   = (__bf16*)take((size_t)GG * HH * 2);
  float*  xpf    = (float*) take((size_t)TT * BB * GG * 4);   // 128 MB
  float*  xpb    = (float*) take((size_t)TT * BB * GG * 4);   // 128 MB

  // 1) fp32 -> bf16 conversions
  const int nx = BB * TT * II;
  cvt_bf16_kernel<<<(nx + 255) / 256, 256, 0, stream>>>(x, xb, nx);
  cvt_bf16_kernel<<<(GG * II + 255) / 256, 256, 0, stream>>>(Wih_f, wihf, GG * II);
  cvt_bf16_kernel<<<(GG * II + 255) / 256, 256, 0, stream>>>(Wih_b, wihb, GG * II);
  cvt_bf16_kernel<<<(GG * HH + 255) / 256, 256, 0, stream>>>(Whh_f, whhf, GG * HH);
  cvt_bf16_kernel<<<(GG * HH + 255) / 256, 256, 0, stream>>>(Whh_b, whhb, GG * HH);

  // 2) zero h/c state + barrier words
  init_state_kernel<<<(BB * HH + 255) / 256, 256, 0, stream>>>(h_f, h_b, c_f, c_b, bar);

  // 3) input projections for both directions (WMMA GEMM, bias folded)
  dim3 gproj(GG / 32, (BB * TT) / 256, 2);
  xproj_kernel<<<gproj, 256, 0, stream>>>(xb, wihf, wihb, b_f, b_b, xpf, xpb);

  // 4) persistent recurrent kernel (both directions concurrently)
  lstm_recurrent_kernel<<<NWG_REC, 256, 0, stream>>>(xpf, xpb, whhf, whhb,
                                                     h_f, h_b, c_f, c_b,
                                                     gts_f, gts_b, out,
                                                     bar, bar + 1);
}
